// ViViT_44384192036973
// MI455X (gfx1250) — matrix operand order repaired and checked by reading, not compiled, not run on hardware
//
#include <hip/hip_runtime.h>
#include <math.h>
#include <stdint.h>

// ---------------------------------------------------------------------------
// ViViT forward for MI455X (gfx1250, wave32, WMMA).
//  * All GEMMs: v_wmma_f32_16x16x32_bf16, bf16 operands stored as bf16
//    (converted once at production), fp32 accumulation, fp32 residual stream.
//  * GEMM is block-cooperative: 8 waves share a 64x32 B tile staged to LDS
//    with gfx1250 async global->LDS copies (double buffered, ASYNCcnt).
//  * V is produced pre-transposed (Vt[feature][token]) so the attention
//    P*V operand is loaded with contiguous b128s instead of strided u16s.
//  * Attention: fused flash-style wave kernel computing S^T = K*Q^T so the
//    softmaxed P tile is A-fragment-native (zero cross-lane transposes).
// ---------------------------------------------------------------------------

typedef __attribute__((ext_vector_type(16))) __bf16 v16bf;
typedef __attribute__((ext_vector_type(8)))  __bf16 v8bf;
typedef __attribute__((ext_vector_type(8)))  float  v8f;

#define DIMD   192
#define HEADS  3
#define DHD    64
#define NTOK   320
#define BTF    16                // B*T frames
#define MTOK   (BTF * NTOK)      // 5120 tokens
#define TT     8

#define USE_ASYNC_COPY 1

static __device__ __forceinline__ __bf16 to_bf(float f) { return (__bf16)f; }

static __device__ __forceinline__ v8f vzero8() {
  v8f z;
#pragma unroll
  for (int i = 0; i < 8; ++i) z[i] = 0.0f;
  return z;
}

static __device__ __forceinline__ v16bf cat16(v8bf lo, v8bf hi) {
  return __builtin_shufflevector(lo, hi, 0, 1, 2, 3, 4, 5, 6, 7, 8, 9, 10, 11,
                                 12, 13, 14, 15);
}

// 16-byte global -> LDS stage. Async path uses the CDNA5 async-tensor data
// path (tracked by ASYNCcnt); fallback is a plain load+ds_store.
static __device__ __forceinline__ void stage16(__bf16* lds_dst, const __bf16* g) {
#if USE_ASYNC_COPY
  uint32_t off = (uint32_t)(uintptr_t)lds_dst;  // low 32 bits = LDS address
  asm volatile("global_load_async_to_lds_b128 %0, %1, off"
               :: "v"(off), "v"(g) : "memory");
#else
  *(v8bf*)lds_dst = *(const v8bf*)g;
#endif
}
static __device__ __forceinline__ void stage_wait() {
#if USE_ASYNC_COPY
  asm volatile("s_wait_asynccnt 0x0" ::: "memory");
#endif
}

// A-fragment (16x32, M rows in-lane): two aligned 16B chunks.
// lanes 0-15 -> M=l ; elems e: K=e+16*half.
static __device__ __forceinline__ v16bf load_a_frag_bf(const __bf16* p, int half) {
  const v8bf* q = (const v8bf*)(p + 16 * half);
  return cat16(q[0], q[1]);  // +0 and +8 elements
}

// ===========================================================================
// GEMM: C[M][N] = A[M][K](bf16) * W[N][K](bf16)^T (+bias)(+gelu)(+res f32)
// Block = 8 waves = 128(M) x 64(N) macro-tile; B tile (64x32 bf16, 4KB)
// staged in LDS (double buffered) and shared by all 8 waves.
// OUTTR stores bf16 transposed: C[N][M] (used for Vt).
// M%128==0, N%64==0, K%32==0 for every call site.
// ===========================================================================
template <bool BIAS, bool GELU, bool RES, bool OUTBF, bool OUTTR>
__global__ void __launch_bounds__(256) gemm_kernel(
    const __bf16* __restrict__ A, const __bf16* __restrict__ W,
    const float* __restrict__ bias, const float* __restrict__ res,
    void* __restrict__ Cout, int M, int N, int K) {
  __shared__ __bf16 sB[2][64 * 32];
  const int tid = threadIdx.x;
  const int wave = tid >> 5, lane = tid & 31;
  const int gm = M >> 7;
  const int bi = blockIdx.x % gm, bj = blockIdx.x / gm;
  const int row0 = bi * 128 + wave * 16, col0 = bj * 64;
  const int nq = lane & 15, half = lane >> 4;

  // Cooperative B staging: thread -> (row 0..63, 16B chunk 0..3).
  const int crow = tid >> 2, cchunk = tid & 3;
  const __bf16* gB = W + (size_t)(col0 + crow) * K + cchunk * 8;
  __bf16* lds0 = &sB[0][crow * 32 + cchunk * 8];
  __bf16* lds1 = &sB[1][crow * 32 + cchunk * 8];

  stage16(lds0, gB);  // tile 0 -> buffer 0

  v8f acc[4];
#pragma unroll
  for (int j = 0; j < 4; ++j) acc[j] = vzero8();

  const __bf16* arow = A + (size_t)(row0 + nq) * K;
  const int steps = K >> 5;
  for (int s = 0; s < steps; ++s) {
    stage_wait();       // own async copies done
    __syncthreads();    // everyone's copies done; prev reads of other buf done
    if (s + 1 < steps) stage16((s & 1) ? lds0 : lds1, gB + (s + 1) * 32);

    v16bf a = load_a_frag_bf(arow + s * 32, half);
    const __bf16* bt = sB[s & 1];
#pragma unroll
    for (int j = 0; j < 4; ++j) {
      // B-fragment: lane -> col n, elems -> K = e + 16*half (contig 16 bf16).
      const v8bf* pb = (const v8bf*)(bt + (j * 16 + nq) * 32 + 16 * half);
      v16bf b = cat16(pb[0], pb[1]);
      acc[j] = __builtin_amdgcn_wmma_f32_16x16x32_bf16(
          false, a, false, b, (short)0, acc[j], false, false);
    }
  }
  // Epilogue. C layout: elem r, lane l -> row = r + 8*half, col = l&15.
#pragma unroll
  for (int j = 0; j < 4; ++j) {
    int cn = col0 + j * 16 + nq;
    float bb = BIAS ? bias[cn] : 0.0f;
#pragma unroll
    for (int r = 0; r < 8; ++r) {
      int rw = row0 + r + 8 * half;
      float v = acc[j][r] + bb;
      if (GELU) v = 0.5f * v * (1.0f + erff(v * 0.70710678118654752f));
      if (OUTTR) {
        ((__bf16*)Cout)[(size_t)cn * M + rw] = to_bf(v);
      } else {
        size_t idx = (size_t)rw * N + cn;
        if (RES) v += res[idx];
        if (OUTBF) ((__bf16*)Cout)[idx] = to_bf(v);
        else       ((float*)Cout)[idx] = v;
      }
    }
  }
}

// ===========================================================================
// Fused flash attention.
//   qk[nseq][seqlen][384]  bf16 (Q | K per head of 64)
//   vt[192][MTOK]          bf16 (V transposed; per-seq token offset s*seqlen)
// One wave per (seq, head, 16-query tile); key loop bounded for the
// block-causal mask (frame_len==seqlen disables masking). Output bf16.
// ===========================================================================
__global__ void attn_kernel(const __bf16* __restrict__ qk,
                            const __bf16* __restrict__ vt,
                            __bf16* __restrict__ O, int nseq, int seqlen,
                            int frame_len) {
  int wave = (blockIdx.x * blockDim.x + threadIdx.x) >> 5;
  int lane = threadIdx.x & 31;
  int qtiles = seqlen >> 4;
  int total = nseq * HEADS * qtiles;
  if (wave >= total) return;
  int qt = wave % qtiles;
  int h = (wave / qtiles) % HEADS;
  int s = wave / (qtiles * HEADS);
  int q0 = qt << 4;
  int kv_limit = ((q0 / frame_len) + 1) * frame_len;
  int nq = lane & 15, half = lane >> 4;
  const __bf16* base = qk + (size_t)s * seqlen * 384;

  // B-fragments of Q^T (features x queries), feature chunks [0,32) and [32,64).
  v16bf bq[2];
  {
    const __bf16* qrow = base + (size_t)(q0 + nq) * 384 + h * DHD;
#pragma unroll
    for (int kk = 0; kk < 2; ++kk) {
      const v8bf* p = (const v8bf*)(qrow + kk * 32 + 16 * half);
      bq[kk] = cat16(p[0], p[1]);
    }
  }

  float run_m = -INFINITY, run_l = 0.0f;
  v8f o[4];
#pragma unroll
  for (int j = 0; j < 4; ++j) o[j] = vzero8();
  const float scale = 0.125f;  // DH^-0.5

  for (int kb = 0; kb < kv_limit; kb += 32) {
    // S^T tiles: st[t] elem r, lane l -> key kb+16t+r+8*half, query l&15.
    v8f st[2];
#pragma unroll
    for (int t = 0; t < 2; ++t) {
      st[t] = vzero8();
      const __bf16* krow = base + (size_t)(kb + t * 16 + nq) * 384 + DIMD + h * DHD;
#pragma unroll
      for (int kk = 0; kk < 2; ++kk) {
        v16bf a = load_a_frag_bf(krow + kk * 32, half);
        st[t] = __builtin_amdgcn_wmma_f32_16x16x32_bf16(
            false, a, false, bq[kk], (short)0, st[t], false, false);
      }
    }
    // Online softmax; lane pair (l, l^16) shares query l&15 -> merge stats.
    float lm = -INFINITY;
#pragma unroll
    for (int r = 0; r < 8; ++r) {
      float a = st[0][r] * scale, b = st[1][r] * scale;
      st[0][r] = a; st[1][r] = b;
      lm = fmaxf(lm, fmaxf(a, b));
    }
    lm = fmaxf(lm, __shfl_xor(lm, 16));
    float new_m = fmaxf(run_m, lm);
    float alpha = __expf(run_m - new_m);
    float av[8];
#pragma unroll
    for (int r = 0; r < 8; ++r) av[r] = __shfl(alpha, r + 8 * half);
#pragma unroll
    for (int j = 0; j < 4; ++j)
#pragma unroll
      for (int r = 0; r < 8; ++r) o[j][r] *= av[r];
    // P is A-fragment-native: e<8 -> st[0][e], e>=8 -> st[1][e-8].
    v16bf p;
    float ls = 0.0f;
#pragma unroll
    for (int e = 0; e < 8; ++e) {
      float v = __expf(st[0][e] - new_m); p[e] = to_bf(v); ls += v;
    }
#pragma unroll
    for (int e = 0; e < 8; ++e) {
      float v = __expf(st[1][e] - new_m); p[8 + e] = to_bf(v); ls += v;
    }
    run_l = run_l * alpha + ls;
    run_m = new_m;
    // O += P * V. V B-fragment: lane -> feature col, elems e<8 -> key 8*half+e, e>=8 -> key 16+8*half+(e-8).
    // Vt is feature-major so each lane loads two 8-token chunks: 2x b128.
#pragma unroll
    for (int j = 0; j < 4; ++j) {
      const __bf16* vrow = vt + (size_t)(h * DHD + j * 16 + nq) * MTOK +
                           (size_t)s * seqlen + kb + 8 * half;
      const v8bf* pv = (const v8bf*)vrow;
      v16bf bv = cat16(pv[0], pv[2]);
      o[j] = __builtin_amdgcn_wmma_f32_16x16x32_bf16(
          false, p, false, bv, (short)0, o[j], false, false);
    }
  }
  run_l += __shfl_xor(run_l, 16);
  float inv[8];
#pragma unroll
  for (int r = 0; r < 8; ++r) inv[r] = 1.0f / __shfl(run_l, r + 8 * half);
  __bf16* orow = O + (size_t)(s * seqlen + q0) * DIMD + h * DHD;
#pragma unroll
  for (int j = 0; j < 4; ++j)
#pragma unroll
    for (int r = 0; r < 8; ++r)
      orow[(size_t)(r + 8 * half) * DIMD + j * 16 + nq] = to_bf(o[j][r] * inv[r]);
}

// ===========================================================================
// LayerNorm over 192 features; one wave per token. OUTBF picks output dtype.
// ===========================================================================
template <bool OUTBF>
__global__ void ln_kernel(const float* __restrict__ X, const float* __restrict__ g,
                          const float* __restrict__ b, void* __restrict__ Y, int M) {
  int wave = (blockIdx.x * blockDim.x + threadIdx.x) >> 5;
  int lane = threadIdx.x & 31;
  if (wave >= M) return;
  const float* x = X + (size_t)wave * DIMD;
  float v[6], s = 0.0f, s2 = 0.0f;
#pragma unroll
  for (int j = 0; j < 6; ++j) {
    v[j] = x[lane + 32 * j];
    s += v[j]; s2 += v[j] * v[j];
  }
#pragma unroll
  for (int off = 16; off; off >>= 1) {
    s += __shfl_xor(s, off);
    s2 += __shfl_xor(s2, off);
  }
  float mean = s * (1.0f / 192.0f);
  float var = s2 * (1.0f / 192.0f) - mean * mean;
  float inv = rsqrtf(var + 1e-5f);
#pragma unroll
  for (int j = 0; j < 6; ++j) {
    int d = lane + 32 * j;
    float y = (v[j] - mean) * inv * g[d] + b[d];
    if (OUTBF) ((__bf16*)Y)[(size_t)wave * DIMD + d] = to_bf(y);
    else       ((float*)Y)[(size_t)wave * DIMD + d] = y;
  }
}

// ===========================================================================
// Data movement / elementwise kernels
// ===========================================================================
__global__ void cvt_kernel(const float* __restrict__ s, __bf16* __restrict__ d, int n) {
  int i = blockIdx.x * blockDim.x + threadIdx.x;
  if (i < n) d[i] = to_bf(s[i]);
}

// im2col 3D patches -> bf16 A3[f*256+p][160], kk order (C3D,PZ,PW,PH).
__global__ void im2col3_kernel(const float* __restrict__ x3d, __bf16* __restrict__ A3) {
  int i = blockIdx.x * blockDim.x + threadIdx.x;
  if (i >= 4096 * 160) return;
  int kk = i % 160, row = i / 160;
  int f = row >> 8, p = row & 255;
  int z = p >> 6, w = (p >> 3) & 7, h = p & 7;
  int c = kk / 32, rem = kk % 32;
  int ii = rem >> 4, j = (rem >> 2) & 3, k = rem & 3;
  size_t src = ((((size_t)f * 8 + (2 * z + ii)) * 32 + (4 * w + j)) * 32 + (4 * h + k)) * 5 + c;
  A3[i] = to_bf(x3d[src]);
}

// im2col 2D patches -> bf16 A2[f*64+p][64], kk order (C2D,PW,PH).
__global__ void im2col2_kernel(const float* __restrict__ x2d, __bf16* __restrict__ A2) {
  int i = blockIdx.x * blockDim.x + threadIdx.x;
  if (i >= 1024 * 64) return;
  int kk = i % 64, row = i / 64;
  int f = row >> 6, p = row & 63;
  int w = p >> 3, h = p & 7;
  int c = kk >> 4, j = (kk >> 2) & 3, k = kk & 3;
  size_t src = (((size_t)f * 32 + (4 * w + j)) * 32 + (4 * h + k)) * 4 + c;
  A2[i] = to_bf(x2d[src]);
}

// X[f][n][d] = PE + pos_emb + temp_emb (all fp32; residual stream stays fp32).
__global__ void assemble_x_kernel(const float* __restrict__ peu,
                                  const float* __restrict__ pes,
                                  const float* __restrict__ pos,
                                  const float* __restrict__ temp,
                                  float* __restrict__ X) {
  int i = blockIdx.x * blockDim.x + threadIdx.x;
  if (i >= MTOK * DIMD) return;
  int d = i % DIMD;
  int n = (i / DIMD) % NTOK;
  int f = i / (DIMD * NTOK);
  int t = f % TT;
  float v = (n < 256) ? peu[((size_t)f * 256 + n) * DIMD + d]
                      : pes[((size_t)f * 64 + (n - 256)) * DIMD + d];
  X[i] = v + pos[n * DIMD + d] + temp[t * DIMD + d];
}

// Gather token slice [tok0, tok0+ntok) per frame into dense bf16 matrix.
__global__ void extract_kernel(const float* __restrict__ X, __bf16* __restrict__ out,
                               int tok0, int ntok) {
  int i = blockIdx.x * blockDim.x + threadIdx.x;
  int total = BTF * ntok * DIMD;
  if (i >= total) return;
  int d = i % DIMD;
  int p = (i / DIMD) % ntok;
  int f = i / (DIMD * ntok);
  out[i] = to_bf(X[((size_t)f * NTOK + tok0 + p) * DIMD + d]);
}

// Transpose + convert: src[K][N] fp32 -> dst[N][K] bf16.
__global__ void transpose_cvt_kernel(const float* __restrict__ src,
                                     __bf16* __restrict__ dst, int K, int N) {
  int i = blockIdx.x * blockDim.x + threadIdx.x;
  if (i >= K * N) return;
  int n = i % N, k = i / N;
  dst[(size_t)n * K + k] = to_bf(src[i]);
}

// Interleave transposed-conv GEMM output (3D): G[4096][2560] -> Y3(16,10,16,64,64)+b
__global__ void permute3_kernel(const float* __restrict__ G, const float* __restrict__ bias,
                                float* __restrict__ Y) {
  int i = blockIdx.x * blockDim.x + threadIdx.x;
  if (i >= 4096 * 2560) return;
  int n = i % 2560, row = i / 2560;
  int f = row >> 8, p = row & 255;
  int z = p >> 6, w = (p >> 3) & 7, h = p & 7;
  int c = n >> 8, ii = (n >> 6) & 3, j = (n >> 3) & 7, k = n & 7;
  size_t idx = ((((size_t)f * 10 + c) * 16 + (z * 4 + ii)) * 64 + (w * 8 + j)) * 64 + (h * 8 + k);
  Y[idx] = G[i] + bias[c];
}

// Interleave transposed-conv GEMM output (2D): G[1024][512] -> Y2(16,8,64,64)+b
__global__ void permute2_kernel(const float* __restrict__ G, const float* __restrict__ bias,
                                float* __restrict__ Y) {
  int i = blockIdx.x * blockDim.x + threadIdx.x;
  if (i >= 1024 * 512) return;
  int n = i % 512, row = i / 512;
  int f = row >> 6, p = row & 63;
  int w = p >> 3, h = p & 7;
  int c = n >> 6, j = (n >> 3) & 7, k = n & 7;
  size_t idx = (((size_t)f * 8 + c) * 64 + (w * 8 + j)) * 64 + (h * 8 + k);
  Y[idx] = G[i] + bias[c];
}

// Refinement conv3d: Y3(16,10,16,64,64), k=3, s=2, p=1 -> out(B,T,Z,W,H,5) ch-last.
__global__ void conv3_kernel(const float* __restrict__ Y, const float* __restrict__ W,
                             const float* __restrict__ bias, float* __restrict__ out) {
  int i = blockIdx.x * blockDim.x + threadIdx.x;
  if (i >= BTF * 8 * 32 * 32 * 5) return;
  int co = i % 5;
  int h = (i / 5) % 32;
  int w = (i / (5 * 32)) % 32;
  int z = (i / (5 * 32 * 32)) % 8;
  int f = i / (5 * 32 * 32 * 8);
  float acc = bias[co];
  for (int ci = 0; ci < 10; ++ci) {
    const float* yb = Y + (((size_t)f * 10 + ci) * 16) * 64 * 64;
    const float* wb = W + ((size_t)co * 10 + ci) * 27;
#pragma unroll
    for (int dz = 0; dz < 3; ++dz) {
      int iz = z * 2 + dz - 1;
      if ((unsigned)iz >= 16u) continue;
#pragma unroll
      for (int dy = 0; dy < 3; ++dy) {
        int iy = w * 2 + dy - 1;
        if ((unsigned)iy >= 64u) continue;
#pragma unroll
        for (int dx = 0; dx < 3; ++dx) {
          int ix = h * 2 + dx - 1;
          if ((unsigned)ix >= 64u) continue;
          acc += yb[((size_t)iz * 64 + iy) * 64 + ix] * wb[(dz * 3 + dy) * 3 + dx];
        }
      }
    }
  }
  out[i] = acc;
}

// Refinement conv2d: Y2(16,8,64,64), k=3, s=2, p=1 -> out(B,T,W,H,4) ch-last.
__global__ void conv2_kernel(const float* __restrict__ Y, const float* __restrict__ W,
                             const float* __restrict__ bias, float* __restrict__ out) {
  int i = blockIdx.x * blockDim.x + threadIdx.x;
  if (i >= BTF * 32 * 32 * 4) return;
  int co = i % 4;
  int h = (i / 4) % 32;
  int w = (i / (4 * 32)) % 32;
  int f = i / (4 * 32 * 32);
  float acc = bias[co];
  for (int ci = 0; ci < 8; ++ci) {
    const float* yb = Y + ((size_t)f * 8 + ci) * 64 * 64;
    const float* wb = W + ((size_t)co * 8 + ci) * 9;
#pragma unroll
    for (int dy = 0; dy < 3; ++dy) {
      int iy = w * 2 + dy - 1;
      if ((unsigned)iy >= 64u) continue;
#pragma unroll
      for (int dx = 0; dx < 3; ++dx) {
        int ix = h * 2 + dx - 1;
        if ((unsigned)ix >= 64u) continue;
        acc += yb[(size_t)iy * 64 + ix] * wb[dy * 3 + dx];
      }
    }
  }
  out[i] = acc;
}

// ===========================================================================
// Host orchestration
// ===========================================================================
// d_in order = jax pytree flatten of setup_inputs(): x_3d, x_2d, then params
// with dict keys sorted alphabetically at every level, lists in index order.
enum {
  IN_X3D = 0, IN_X2D = 1,
  P_PE_CS_B = 2, P_PE_CS_W = 3, P_PE_CU_B = 4, P_PE_CU_W = 5, P_POS_EMB = 6,
  P_RFS_B = 7, P_RFS_W = 8, P_RFU_B = 9, P_RFU_W = 10,
  P_SPACE = 11,                       // 4 layers x 11 leaves
  P_SPACE_NB = 55, P_SPACE_NG = 56,
  P_TCS_B = 57, P_TCS_W = 58, P_TCU_B = 59, P_TCU_W = 60,
  P_TEMP_EMB = 61, P_TEMP_NB = 62, P_TEMP_NG = 63,
  P_TEMPORAL = 64,                    // 4 layers x 11 leaves
};
// layer leaf order (sorted): b1,b2,b_out,ln1_b,ln1_g,ln2_b,ln2_g,w1,w2,w_out,w_qkv

struct Layer {
  const float *b1, *b2, *b_out, *ln1_b, *ln1_g, *ln2_b, *ln2_g, *w1, *w2, *w_out, *w_qkv;
};

#define NQKV 110592   // 576*192
#define NOUT 36864    // 192*192
#define NW1  147456   // 768*192
#define NW2  147456   // 192*768
#define NLW  (NQKV + NOUT + NW1 + NW2)

extern "C" void kernel_launch(void* const* d_in, const int* in_sizes, int n_in,
                              void* d_out, int out_size, void* d_ws, size_t ws_size,
                              hipStream_t stream) {
  (void)in_sizes; (void)n_in; (void)out_size; (void)ws_size;
  auto F = [&](int i) { return (const float*)d_in[i]; };
  auto layer = [&](int base) {
    Layer L;
    L.b1 = F(base + 0);  L.b2 = F(base + 1);  L.b_out = F(base + 2);
    L.ln1_b = F(base + 3); L.ln1_g = F(base + 4);
    L.ln2_b = F(base + 5); L.ln2_g = F(base + 6);
    L.w1 = F(base + 7); L.w2 = F(base + 8); L.w_out = F(base + 9); L.w_qkv = F(base + 10);
    return L;
  };

  // Workspace (byte offsets, 256B aligned). Total ~125 MB.
  char* base = (char*)d_ws;
  size_t off = 0;
  auto take = [&](size_t bytes) {
    char* p = base + off;
    off += (bytes + 255) & ~(size_t)255;
    return p;
  };
  float*  X    = (float*)take((size_t)MTOK * DIMD * 4);
  __bf16* XN   = (__bf16*)take((size_t)MTOK * DIMD * 2);
  __bf16* QK   = (__bf16*)take((size_t)MTOK * 2 * DIMD * 2);  // [tok][384]
  __bf16* Vt   = (__bf16*)take((size_t)DIMD * MTOK * 2);      // [feat][tok]
  __bf16* Oa   = (__bf16*)take((size_t)MTOK * DIMD * 2);
  __bf16* Hb   = (__bf16*)take((size_t)MTOK * 4 * DIMD * 2);
  __bf16* A3   = (__bf16*)take(4096ull * 160 * 2);
  __bf16* A2   = (__bf16*)take(1024ull * 64 * 2);
  float*  PEu  = (float*)take(4096ull * DIMD * 4);
  float*  PEs  = (float*)take(1024ull * DIMD * 4);
  __bf16* WuT  = (__bf16*)take(2560ull * DIMD * 2);
  __bf16* WsT  = (__bf16*)take(512ull * DIMD * 2);
  __bf16* Xu   = (__bf16*)take(4096ull * DIMD * 2);
  __bf16* Xs   = (__bf16*)take(1024ull * DIMD * 2);
  float*  Gu   = (float*)take(4096ull * 2560 * 4);
  float*  Gs   = (float*)take(1024ull * 512 * 4);
  float*  Y3   = (float*)take(16ull * 10 * 16 * 64 * 64 * 4);
  float*  Y2   = (float*)take(16ull * 8 * 64 * 64 * 4);
  __bf16* WL   = (__bf16*)take(8ull * NLW * 2);   // bf16 layer weights
  __bf16* PW3  = (__bf16*)take(30720ull * 2);     // pe_cu_w bf16
  __bf16* PW2  = (__bf16*)take(12288ull * 2);     // pe_cs_w bf16

  float* out_up = (float*)d_out;            // (2,8,8,32,32,5)
  float* out_sf = (float*)d_out + 655360;   // (2,8,32,32,4)

  dim3 blk(256);
  auto grid1 = [&](size_t total) { return dim3((unsigned)((total + 255) / 256)); };
  auto cvt = [&](const float* s, __bf16* d, int n) {
    cvt_kernel<<<grid1(n), blk, 0, stream>>>(s, d, n);
  };
  // Convert all transformer weights to bf16 once.
  for (int st = 0; st < 2; ++st) {
    for (int l = 0; l < 4; ++l) {
      Layer L = layer((st ? P_TEMPORAL : P_SPACE) + l * 11);
      __bf16* lw = WL + (size_t)(st * 4 + l) * NLW;
      cvt(L.w_qkv, lw, NQKV);
      cvt(L.w_out, lw + NQKV, NOUT);
      cvt(L.w1,    lw + NQKV + NOUT, NW1);
      cvt(L.w2,    lw + NQKV + NOUT + NW1, NW2);
    }
  }
  cvt(F(P_PE_CU_W), PW3, 30720);
  cvt(F(P_PE_CS_W), PW2, 12288);
  transpose_cvt_kernel<<<grid1(2560ull * DIMD), blk, 0, stream>>>(F(P_TCU_W), WuT, DIMD, 2560);
  transpose_cvt_kernel<<<grid1(512ull * DIMD), blk, 0, stream>>>(F(P_TCS_W), WsT, DIMD, 512);

  auto gemm = [&](int mode, const __bf16* A, const __bf16* Wm, const float* bias,
                  const float* res, void* C, int M, int N, int K) {
    dim3 g((M >> 7) * (N >> 6));
    switch (mode) {
      case 0: gemm_kernel<true,  false, false, false, false><<<g, blk, 0, stream>>>(A, Wm, bias, res, C, M, N, K); break; // PE
      case 1: gemm_kernel<false, false, false, true,  false><<<g, blk, 0, stream>>>(A, Wm, bias, res, C, M, N, K); break; // QK
      case 2: gemm_kernel<true,  false, true,  false, false><<<g, blk, 0, stream>>>(A, Wm, bias, res, C, M, N, K); break; // proj+res
      case 3: gemm_kernel<true,  true,  false, true,  false><<<g, blk, 0, stream>>>(A, Wm, bias, res, C, M, N, K); break; // mlp1
      case 4: gemm_kernel<false, false, false, false, false><<<g, blk, 0, stream>>>(A, Wm, bias, res, C, M, N, K); break; // recovery
      case 5: gemm_kernel<false, false, false, true,  true ><<<g, blk, 0, stream>>>(A, Wm, bias, res, C, M, N, K); break; // V transposed
    }
  };

  // ---- patch embedding ----
  im2col3_kernel<<<grid1(4096ull * 160), blk, 0, stream>>>(F(IN_X3D), A3);
  im2col2_kernel<<<grid1(1024ull * 64), blk, 0, stream>>>(F(IN_X2D), A2);
  gemm(0, A3, PW3, F(P_PE_CU_B), nullptr, PEu, 4096, DIMD, 160);
  gemm(0, A2, PW2, F(P_PE_CS_B), nullptr, PEs, 1024, DIMD, 64);
  assemble_x_kernel<<<grid1((size_t)MTOK * DIMD), blk, 0, stream>>>(
      PEu, PEs, F(P_POS_EMB), F(P_TEMP_EMB), X);

  // ---- transformer stacks ----
  auto run_stack = [&](int stackIdx, int pbase, const float* ng, const float* nb,
                       int nseq, int seqlen, int frame_len) {
    for (int l = 0; l < 4; ++l) {
      Layer L = layer(pbase + l * 11);
      const __bf16* lw = WL + (size_t)(stackIdx * 4 + l) * NLW;
      ln_kernel<true><<<dim3(MTOK / 8), blk, 0, stream>>>(X, L.ln1_g, L.ln1_b, XN, MTOK);
      gemm(1, XN, lw, nullptr, nullptr, QK, MTOK, 2 * DIMD, DIMD);           // Q|K rows 0..383
      gemm(5, XN, lw + (size_t)2 * DIMD * DIMD, nullptr, nullptr, Vt,
           MTOK, DIMD, DIMD);                                                // V rows 384..575 -> Vt
      {
        int waves = nseq * HEADS * (seqlen >> 4);
        attn_kernel<<<dim3((waves + 7) / 8), blk, 0, stream>>>(QK, Vt, Oa, nseq, seqlen, frame_len);
      }
      gemm(2, Oa, lw + NQKV, L.b_out, X, X, MTOK, DIMD, DIMD);
      ln_kernel<true><<<dim3(MTOK / 8), blk, 0, stream>>>(X, L.ln2_g, L.ln2_b, XN, MTOK);
      gemm(3, XN, lw + NQKV + NOUT, L.b1, nullptr, Hb, MTOK, 4 * DIMD, DIMD);
      gemm(2, Hb, lw + NQKV + NOUT + NW1, L.b2, X, X, MTOK, DIMD, 4 * DIMD);
    }
    ln_kernel<false><<<dim3(MTOK / 8), blk, 0, stream>>>(X, ng, nb, X, MTOK);
  };
  run_stack(0, P_SPACE, F(P_SPACE_NG), F(P_SPACE_NB), BTF, NTOK, NTOK);        // spatial
  run_stack(1, P_TEMPORAL, F(P_TEMP_NG), F(P_TEMP_NB), 2, TT * NTOK, NTOK);    // temporal

  // ---- patch recovery ----
  extract_kernel<<<grid1(4096ull * DIMD), blk, 0, stream>>>(X, Xu, 0, 256);
  extract_kernel<<<grid1(1024ull * DIMD), blk, 0, stream>>>(X, Xs, 256, 64);
  gemm(4, Xu, WuT, nullptr, nullptr, Gu, 4096, 2560, DIMD);
  gemm(4, Xs, WsT, nullptr, nullptr, Gs, 1024, 512, DIMD);
  permute3_kernel<<<grid1(4096ull * 2560), blk, 0, stream>>>(Gu, F(P_TCU_B), Y3);
  permute2_kernel<<<grid1(1024ull * 512), blk, 0, stream>>>(Gs, F(P_TCS_B), Y2);
  conv3_kernel<<<grid1((size_t)BTF * 8 * 32 * 32 * 5), blk, 0, stream>>>(Y3, F(P_RFU_W), F(P_RFU_B), out_up);
  conv2_kernel<<<grid1((size_t)BTF * 32 * 32 * 4), blk, 0, stream>>>(Y2, F(P_RFS_W), F(P_RFS_B), out_sf);
}
